// QNetwork_14216341750447
// MI455X (gfx1250) — matrix operand order repaired and checked by reading, not compiled, not run on hardware
//
#include <hip/hip_runtime.h>
#include <hip/hip_bf16.h>

#define TT 128
#define BB 256
#define FF 512
#define HH 512
#define AA 16
#define LL 4

typedef __attribute__((ext_vector_type(16))) __bf16 v16bf;
typedef __attribute__((ext_vector_type(8)))  float  v8f;
typedef __attribute__((ext_vector_type(4)))  unsigned int u32x4;

union BfFrag { v16bf v; u32x4 u[2]; };

// A fragment: 16x32 bf16, row-major source. Per ISA: lanes0-15 rows M=0..15 hold
// K {kg*8..kg*8+7, 16+kg*8..}, kg = lane>>4.
__device__ __forceinline__ v16bf load_a_frag(const __bf16* A, int lda, int m0, int k0, unsigned lane) {
  int m  = m0 + (int)(lane & 15u);
  int kg = (int)(lane >> 4) * 8;
  const __bf16* p = A + (size_t)m * lda + (size_t)(k0 + kg);
  BfFrag f;
  f.u[0] = *(const u32x4*)(p);
  f.u[1] = *(const u32x4*)(p + 16);
  return f.v;
}

// B fragment: 32x16 bf16 (KxN), loaded from pre-transposed weight Bt (N x K row-major).
// Lane l: N = l&15, K {kg*8..kg*8+7, 16+kg*8..}, kg = l>>4.
__device__ __forceinline__ v16bf load_bt_frag(const __bf16* Bt, int ldb, int n0, int k0, unsigned lane) {
  int n  = n0 + (int)(lane & 15u);
  int kg = (int)(lane >> 4) * 8;
  const __bf16* p = Bt + (size_t)n * ldb + (size_t)(k0 + kg);
  BfFrag f;
  f.u[0] = *(const u32x4*)(p);
  f.u[1] = *(const u32x4*)(p + 16);
  return f.v;
}

#define WMMA_BF16(a, b, c) \
  __builtin_amdgcn_wmma_f32_16x16x32_bf16(false, (a), false, (b), (short)0, (c), false, false)

// ---------------------------------------------------------------- conversions

__global__ __launch_bounds__(256) void f32_to_bf16_kernel(const float* __restrict__ in,
                                                          __bf16* __restrict__ out, int n) {
  int i = blockIdx.x * 256 + threadIdx.x;
  if (i < n) out[i] = (__bf16)in[i];
}

// W_mlp (L,H,H) row-major -> WmlpT (L, N=H, K=H): WmlpT[l][n][k] = W[l][k][n]
__global__ __launch_bounds__(256) void transpose_wmlp_kernel(const float* __restrict__ W,
                                                             __bf16* __restrict__ Wt, int n) {
  int i = blockIdx.x * 256 + threadIdx.x;
  if (i >= n) return;
  int l = i / (HH * HH);
  int r = i % (HH * HH);
  int nn = r / HH;
  int k  = r % HH;
  Wt[i] = (__bf16)W[(size_t)l * HH * HH + (size_t)k * HH + nn];
}

// Wi/Wh (H, 4H) -> permuted-transposed (4H, H): row p = unit*4+gate maps to
// original column gate*H + unit.
__global__ __launch_bounds__(256) void permute_w_kernel(const float* __restrict__ W,
                                                        __bf16* __restrict__ Wt, int n) {
  int i = blockIdx.x * 256 + threadIdx.x;
  if (i >= n) return;
  int p = i / HH;
  int k = i % HH;
  int gate = p & 3, u = p >> 2;
  Wt[i] = (__bf16)W[(size_t)k * (4 * HH) + (size_t)gate * HH + u];
}

__global__ __launch_bounds__(256) void permute_bias_kernel(const float* __restrict__ b,
                                                           float* __restrict__ bp) {
  int p = blockIdx.x * 256 + threadIdx.x;
  if (p < 4 * HH) bp[p] = b[(p & 3) * HH + (p >> 2)];
}

__global__ __launch_bounds__(256) void init_state_kernel(const float* __restrict__ hc,
                                                         const float* __restrict__ hh,
                                                         float* __restrict__ C,
                                                         __bf16* __restrict__ hb) {
  int i = blockIdx.x * 256 + threadIdx.x;
  if (i < BB * HH) {
    C[i]  = hc[i];
    hb[i] = (__bf16)hh[i];
  }
}

// ------------------------------------------------- fused MLP layer (GEMM+LN+ReLU)
// Block: 256 threads = 8 waves; tile = 16 rows x full 512 cols (wave w -> cols w*64..+63).
__global__ __launch_bounds__(256) void mlp_layer_kernel(const __bf16* __restrict__ X,
                                                        const __bf16* __restrict__ Wt,
                                                        const float* __restrict__ bias,
                                                        const float* __restrict__ gamma,
                                                        const float* __restrict__ beta,
                                                        __bf16* __restrict__ Y) {
  int m0 = blockIdx.x * 16;
  unsigned tid  = threadIdx.x;
  unsigned lane = tid & 31u;
  unsigned wid  = tid >> 5;
  int nbase = (int)wid * 64;

  v8f acc[4] = {};
  for (int k0 = 0; k0 < HH; k0 += 32) {
    v16bf a = load_a_frag(X, HH, m0, k0, lane);
#pragma unroll
    for (int f = 0; f < 4; ++f) {
      v16bf b = load_bt_frag(Wt, HH, nbase + f * 16, k0, lane);
      acc[f] = WMMA_BF16(a, b, acc[f]);
    }
  }

  int rbase = (int)(lane >> 4) * 8;
  int ncol  = (int)(lane & 15u);
  // add bias before LN stats
#pragma unroll
  for (int f = 0; f < 4; ++f) {
    float bb = bias[nbase + f * 16 + ncol];
#pragma unroll
    for (int r = 0; r < 8; ++r) acc[f][r] = acc[f][r] + bb;
  }

  __shared__ float s_sum[8][16];
  __shared__ float s_sq[8][16];
#pragma unroll
  for (int r = 0; r < 8; ++r) {
    float s = 0.f, q = 0.f;
#pragma unroll
    for (int f = 0; f < 4; ++f) { float v = acc[f][r]; s += v; q += v * v; }
#pragma unroll
    for (int m = 1; m <= 8; m <<= 1) {   // reduce across the 16-lane half-wave
      s += __shfl_xor(s, m, 32);
      q += __shfl_xor(q, m, 32);
    }
    if ((lane & 15u) == 0u) { s_sum[wid][rbase + r] = s; s_sq[wid][rbase + r] = q; }
  }
  __syncthreads();

#pragma unroll
  for (int r = 0; r < 8; ++r) {
    int row = rbase + r;
    float S = 0.f, Q = 0.f;
#pragma unroll
    for (int w = 0; w < 8; ++w) { S += s_sum[w][row]; Q += s_sq[w][row]; }
    float mu  = S * (1.0f / 512.0f);
    float var = Q * (1.0f / 512.0f) - mu * mu;
    float inv = rsqrtf(var + 1e-6f);
#pragma unroll
    for (int f = 0; f < 4; ++f) {
      int n = nbase + f * 16 + ncol;
      float v = (acc[f][r] - mu) * inv * gamma[n] + beta[n];
      v = v > 0.f ? v : 0.f;
      Y[(size_t)(m0 + row) * HH + n] = (__bf16)v;
    }
  }
}

// --------------------------------------------- bulk gates_x = X @ WiT(perm) + b
// Block: 8 waves, tile M=32 x N=256 (wave: (wid>>2) M-tile, (wid&3) 64-col strip).
__global__ __launch_bounds__(256) void gemm_gates_x_kernel(const __bf16* __restrict__ X,
                                                           const __bf16* __restrict__ WiT,
                                                           const float* __restrict__ bperm,
                                                           float* __restrict__ G) {
  const int N = 4 * HH;
  int m0 = blockIdx.y * 32;
  int n0 = blockIdx.x * 256;
  unsigned tid  = threadIdx.x;
  unsigned lane = tid & 31u;
  unsigned wid  = tid >> 5;
  int mw = m0 + (int)(wid >> 2) * 16;
  int nw = n0 + (int)(wid & 3) * 64;

  v8f acc[4] = {};
  for (int k0 = 0; k0 < HH; k0 += 32) {
    v16bf a = load_a_frag(X, HH, mw, k0, lane);
#pragma unroll
    for (int f = 0; f < 4; ++f) {
      v16bf b = load_bt_frag(WiT, HH, nw + f * 16, k0, lane);
      acc[f] = WMMA_BF16(a, b, acc[f]);
    }
  }

  int rbase = (int)(lane >> 4) * 8;
  int ncol  = (int)(lane & 15u);
#pragma unroll
  for (int f = 0; f < 4; ++f) {
    int n = nw + f * 16 + ncol;
    float bb = bperm[n];
#pragma unroll
    for (int r = 0; r < 8; ++r)
      G[(size_t)(mw + rbase + r) * N + n] = acc[f][r] + bb;
  }
}

// ----------------------------------------------------------- LSTM step (one t)
// Block: 128 threads = 4 waves; tile M=16 batch rows x N=256 permuted gate cols
// (= 64 hidden units, all 4 gates). Fused cell update via LDS.
__global__ __launch_bounds__(128) void lstm_step_kernel(const __bf16* __restrict__ Hb,
                                                        const __bf16* __restrict__ WhT,
                                                        const float* __restrict__ Gx,
                                                        const int* __restrict__ dones_t,
                                                        float* __restrict__ C,
                                                        float* __restrict__ Yt,
                                                        __bf16* __restrict__ HbOut) {
  const int N = 4 * HH;
  int m0 = blockIdx.y * 16;
  int n0 = blockIdx.x * 256;
  unsigned tid  = threadIdx.x;
  unsigned lane = tid & 31u;
  unsigned wid  = tid >> 5;   // 0..3
  int nw = n0 + (int)wid * 64;

  int arow = m0 + (int)(lane & 15u);
  bool adone = dones_t[arow] != 0;

  v8f acc[4] = {};
  for (int k0 = 0; k0 < HH; k0 += 32) {
    BfFrag af;
    if (adone) {
      u32x4 z = {0u, 0u, 0u, 0u};
      af.u[0] = z; af.u[1] = z;
    } else {
      int kg = (int)(lane >> 4) * 8;
      const __bf16* p = Hb + (size_t)arow * HH + (size_t)(k0 + kg);
      af.u[0] = *(const u32x4*)(p);
      af.u[1] = *(const u32x4*)(p + 16);
    }
#pragma unroll
    for (int f = 0; f < 4; ++f) {
      v16bf b = load_bt_frag(WhT, HH, nw + f * 16, k0, lane);
      acc[f] = WMMA_BF16(af.v, b, acc[f]);
    }
  }

  __shared__ float ldsG[16][256];
  int rbase = (int)(lane >> 4) * 8;
  int ncol  = (int)(lane & 15u);
#pragma unroll
  for (int f = 0; f < 4; ++f) {
    int nloc = (int)wid * 64 + f * 16 + ncol;
    int n = n0 + nloc;
#pragma unroll
    for (int r = 0; r < 8; ++r)
      ldsG[rbase + r][nloc] = acc[f][r] + Gx[(size_t)(m0 + rbase + r) * N + n];
  }
  __syncthreads();

  int u0 = n0 >> 2;  // global hidden-unit base
  for (int e = (int)tid; e < 16 * 64; e += 128) {
    int row = e >> 6;
    int u   = e & 63;
    float gi = ldsG[row][u * 4 + 0];
    float gf = ldsG[row][u * 4 + 1];
    float gg = ldsG[row][u * 4 + 2];
    float go = ldsG[row][u * 4 + 3];
    int b  = m0 + row;
    int hu = u0 + u;
    float c_old = (dones_t[b] != 0) ? 0.f : C[(size_t)b * HH + hu];
    float si = 1.f / (1.f + expf(-gi));
    float sf = 1.f / (1.f + expf(-gf));
    float so = 1.f / (1.f + expf(-go));
    float nc = sf * c_old + si * tanhf(gg);
    float nh = so * tanhf(nc);
    C[(size_t)b * HH + hu]     = nc;
    Yt[(size_t)b * HH + hu]    = nh;
    HbOut[(size_t)b * HH + hu] = (__bf16)nh;
  }
}

// ----------------------------------------------------------------- dueling head
__global__ __launch_bounds__(256) void heads_kernel(const float* __restrict__ Ys,
                                                    const float* __restrict__ W_adv,
                                                    const float* __restrict__ b_adv,
                                                    const float* __restrict__ W_val,
                                                    const float* __restrict__ b_val,
                                                    float* __restrict__ Q) {
  int row = blockIdx.x * 8 + (int)(threadIdx.x >> 5);
  unsigned lane = threadIdx.x & 31u;
  const float* h = Ys + (size_t)row * HH;

  float hv[16];
#pragma unroll
  for (int j = 0; j < 16; ++j) hv[j] = h[lane + 32 * j];

  float adv[AA];
#pragma unroll
  for (int a = 0; a < AA; ++a) {
    float p = 0.f;
#pragma unroll
    for (int j = 0; j < 16; ++j) p += hv[j] * W_adv[(size_t)(lane + 32 * j) * AA + a];
#pragma unroll
    for (int m = 1; m <= 16; m <<= 1) p += __shfl_xor(p, m, 32);
    adv[a] = p + b_adv[a];
  }
  float vv = 0.f;
#pragma unroll
  for (int j = 0; j < 16; ++j) vv += hv[j] * W_val[lane + 32 * j];
#pragma unroll
  for (int m = 1; m <= 16; m <<= 1) vv += __shfl_xor(vv, m, 32);
  vv += b_val[0];

  float mean = 0.f;
#pragma unroll
  for (int a = 0; a < AA; ++a) mean += adv[a];
  mean *= (1.0f / 16.0f);

  if (lane == 0) {
#pragma unroll
    for (int a = 0; a < AA; ++a) Q[(size_t)row * AA + a] = vv + adv[a] - mean;
  }
}

__global__ __launch_bounds__(256) void copyout_kernel(const float* __restrict__ C,
                                                      const float* __restrict__ Hlast,
                                                      float* __restrict__ out) {
  int i = blockIdx.x * 256 + threadIdx.x;
  if (i < BB * HH) {
    out[i]           = C[i];
    out[BB * HH + i] = Hlast[i];
  }
}

// --------------------------------------------------------------------- driver

extern "C" void kernel_launch(void* const* d_in, const int* in_sizes, int n_in,
                              void* d_out, int out_size, void* d_ws, size_t ws_size,
                              hipStream_t stream) {
  const float* hidden_c = (const float*)d_in[0];
  const float* hidden_h = (const float*)d_in[1];
  const float* x        = (const float*)d_in[2];
  const int*   dones    = (const int*)d_in[3];
  const float* W_mlp    = (const float*)d_in[4];
  const float* b_mlp    = (const float*)d_in[5];
  const float* g_mlp    = (const float*)d_in[6];
  const float* be_mlp   = (const float*)d_in[7];
  const float* Wi       = (const float*)d_in[8];
  const float* Wh       = (const float*)d_in[9];
  const float* b_lstm   = (const float*)d_in[10];
  const float* W_adv    = (const float*)d_in[11];
  const float* b_adv    = (const float*)d_in[12];
  const float* W_val    = (const float*)d_in[13];
  const float* b_val    = (const float*)d_in[14];
  float* out = (float*)d_out;
  (void)in_sizes; (void)n_in; (void)out_size; (void)ws_size;

  char* base = (char*)d_ws;
  size_t off = 0;
  auto take = [&](size_t bytes) -> char* {
    char* p = base + off;
    off += (bytes + 255) & ~(size_t)255;
    return p;
  };

  const size_t M = (size_t)TT * BB;  // 32768 rows
  __bf16* xb0   = (__bf16*)take(M * HH * 2);
  __bf16* xb1   = (__bf16*)take(M * HH * 2);
  __bf16* WmlpT = (__bf16*)take((size_t)LL * HH * HH * 2);
  __bf16* WiT   = (__bf16*)take((size_t)4 * HH * HH * 2);
  __bf16* WhT   = (__bf16*)take((size_t)4 * HH * HH * 2);
  float*  bperm = (float*)take((size_t)4 * HH * 4);
  float*  G     = (float*)take(M * 4 * HH * 4);       // (T,B,4H) permuted gates_x
  float*  Ys    = (float*)take(M * HH * 4);           // (T,B,H) hidden outputs
  float*  C     = (float*)take((size_t)BB * HH * 4);  // cell state
  __bf16* hb0   = (__bf16*)take((size_t)BB * HH * 2);
  __bf16* hb1   = (__bf16*)take((size_t)BB * HH * 2);
  __bf16* hb[2] = {hb0, hb1};

  // 1) conversions / permutations
  {
    int n = (int)(M * HH);
    f32_to_bf16_kernel<<<(n + 255) / 256, 256, 0, stream>>>(x, xb0, n);
  }
  {
    int n = LL * HH * HH;
    transpose_wmlp_kernel<<<(n + 255) / 256, 256, 0, stream>>>(W_mlp, WmlpT, n);
  }
  {
    int n = 4 * HH * HH;
    permute_w_kernel<<<(n + 255) / 256, 256, 0, stream>>>(Wi, WiT, n);
    permute_w_kernel<<<(n + 255) / 256, 256, 0, stream>>>(Wh, WhT, n);
  }
  permute_bias_kernel<<<(4 * HH + 255) / 256, 256, 0, stream>>>(b_lstm, bperm);
  init_state_kernel<<<(BB * HH + 255) / 256, 256, 0, stream>>>(hidden_c, hidden_h, C, hb[0]);

  // 2) MLP stack (ping-pong bf16 buffers; ends in xb0)
  __bf16* bufs[2] = {xb0, xb1};
  for (int l = 0; l < LL; ++l) {
    mlp_layer_kernel<<<(int)(M / 16), 256, 0, stream>>>(
        bufs[l & 1], WmlpT + (size_t)l * HH * HH,
        b_mlp + (size_t)l * HH, g_mlp + (size_t)l * HH, be_mlp + (size_t)l * HH,
        bufs[(l + 1) & 1]);
  }

  // 3) bulk input projection: G = mlp_out @ WiT(perm) + bperm
  {
    dim3 grid(4 * HH / 256, (unsigned)(M / 32));
    gemm_gates_x_kernel<<<grid, 256, 0, stream>>>(xb0, WiT, bperm, G);
  }

  // 4) recurrence
  for (int t = 0; t < TT; ++t) {
    dim3 grid(4 * HH / 256, BB / 16);
    lstm_step_kernel<<<grid, 128, 0, stream>>>(
        hb[t & 1], WhT,
        G + (size_t)t * BB * 4 * HH,
        dones + (size_t)t * BB,
        C,
        Ys + (size_t)t * BB * HH,
        hb[(t + 1) & 1]);
  }

  // 5) outputs
  copyout_kernel<<<(BB * HH + 255) / 256, 256, 0, stream>>>(
      C, Ys + (size_t)(TT - 1) * BB * HH, out);
  heads_kernel<<<(int)(M / 8), 256, 0, stream>>>(
      Ys, W_adv, b_adv, W_val, b_val, out + 2 * (size_t)BB * HH);
}
